// E4_C4_32658931318949
// MI455X (gfx1250) — matrix operand order repaired and checked by reading, not compiled, not run on hardware
//
#include <hip/hip_runtime.h>

typedef __attribute__((ext_vector_type(16))) __bf16 v16bf;
typedef __attribute__((ext_vector_type(8)))  float  v8f;

__device__ __forceinline__ unsigned short f32_to_bf16(float f) {
    unsigned int u = __float_as_uint(f);
    unsigned int lsb = (u >> 16) & 1u;
    u += 0x7fffu + lsb;                 // round to nearest even
    return (unsigned short)(u >> 16);
}

// ---------------------------------------------------------------- converts
__global__ __launch_bounds__(256)
void cvt_bf16_kernel(const float* __restrict__ in, unsigned short* __restrict__ out, int n) {
    int i = blockIdx.x * 256 + threadIdx.x;
    if (i < n) out[i] = f32_to_bf16(in[i]);
}

// C4 lift: W[o*4+r, i*4+rr] = w[o, i, (rr-r)&3], cols = 128*4 = 512
__global__ __launch_bounds__(256)
void lift_kernel(const float* __restrict__ w, unsigned short* __restrict__ out, int total) {
    int t = blockIdx.x * 256 + threadIdx.x;
    if (t >= total) return;
    int col = t & 511;
    int row = t >> 9;
    int o = row >> 2, r = row & 3;
    int i = col >> 2, rr = col & 3;
    out[t] = f32_to_bf16(w[(o * 128 + i) * 4 + ((rr - r) & 3)]);
}

// ------------------------------------------------- bf16 WMMA GEMM
// Block = 8 waves, one M-tile (16 rows) per block staged in LDS via async copy,
// each wave computes a 16x32 strip (2 N-tiles).  C = A*B (+bias), N = Ntiles*16.
__global__ __launch_bounds__(256)
void wmma_gemm_bf16(const unsigned short* __restrict__ A,
                    const unsigned short* __restrict__ B,
                    float* __restrict__ C,
                    const float* __restrict__ bias,
                    int M, int Mtiles, int Ntiles, int K, int lda,
                    long strideB, long strideC) {
    __shared__ __align__(16) unsigned short As[16 * 512];   // 16 rows x K (K<=512)

    const int nG = Ntiles >> 4;                 // 16 N-tiles per block
    int ng  = blockIdx.x % nG;
    int tmp = blockIdx.x / nG;
    int mt  = tmp % Mtiles;
    int b   = tmp / Mtiles;
    const int N = Ntiles << 4;

    // ---- stage A strip [16 x K] into LDS with async-to-LDS (ASYNCcnt path)
    const int chunksPerRow = K >> 3;            // 16B chunks per row
    const int totalChunks  = chunksPerRow << 4; // 16 rows
    for (int t = threadIdx.x; t < totalChunks; t += 256) {
        int row = t / chunksPerRow;
        int cc  = t - row * chunksPerRow;
        int gm  = mt * 16 + row; if (gm > M - 1) gm = M - 1;   // clamp; rows >= M not stored
        const unsigned short* gp = A + (long)gm * lda + (cc << 3);
        unsigned loff = (unsigned)(size_t)(&As[row * K + (cc << 3)]);  // low 32b = LDS offset
        asm volatile("global_load_async_to_lds_b128 %0, %1, off"
                     :: "v"(loff), "v"(gp) : "memory");
    }
    asm volatile("s_wait_asynccnt 0x0" ::: "memory");
    __syncthreads();

    const int wave = threadIdx.x >> 5;
    const int lane = threadIdx.x & 31;
    const int half = lane >> 4;
    const int l16  = lane & 15;
    const int nt0  = ng * 16 + wave * 2;        // this wave's two N-tiles

    // A fragment source in LDS: row M = l16, element i holds K = (i<8?0:16)+half*8+(i&7)
    const unsigned short* Arow  = &As[l16 * K + half * 8];
    // B fragment: column N = nt0*16 + l16 (+16), element i holds K = (i<8?0:16)+half*8+(i&7)
    const unsigned short* Bbase = B + (long)b * strideB + nt0 * 16 + l16;
    float* Cbase = C + (long)b * strideC;

    v8f acc0 = {0.f, 0.f, 0.f, 0.f, 0.f, 0.f, 0.f, 0.f};
    v8f acc1 = acc0;
    for (int k0 = 0; k0 < K; k0 += 32) {
        union { uint4 q[2]; v16bf v; unsigned short e[16]; } a, b0, b1;
        a.q[0] = *reinterpret_cast<const uint4*>(Arow + k0);        // ds_load_b128
        a.q[1] = *reinterpret_cast<const uint4*>(Arow + k0 + 16);
        const unsigned short* bp = Bbase + (long)k0 * N;
        for (int i = 0; i < 16; ++i) {
            int kk = (i < 8 ? 0 : 16) + half * 8 + (i & 7);
            b0.e[i] = bp[(long)kk * N];
            b1.e[i] = bp[(long)kk * N + 16];
        }
        __builtin_prefetch(bp + 32 * N, 0, 1);                      // global_prefetch_b8 (next K-tile)
        acc0 = __builtin_amdgcn_wmma_f32_16x16x32_bf16(
            false, a.v, false, b0.v, (short)0, acc0, false, false);
        acc1 = __builtin_amdgcn_wmma_f32_16x16x32_bf16(
            false, a.v, false, b1.v, (short)0, acc1, false, false);
    }

    int n0 = nt0 * 16 + l16;
#pragma unroll
    for (int r = 0; r < 8; ++r) {
        int m = mt * 16 + half * 8 + r;
        if (m < M) {
            float bv = bias ? bias[m] : 0.f;
            Cbase[(long)m * N + n0]      = acc0[r] + bv;
            Cbase[(long)m * N + n0 + 16] = acc1[r] + bv;
        }
    }
}

// ------------------------------------------------- GroupNorm(4ch) + ReLU -> bf16
// t: [B,256,1024] f32 ; tn: [B,64,4096] bf16 (K-major for conv2 GEMM)
__global__ __launch_bounds__(256)
void gn_relu_kernel(const float* __restrict__ t, const float* __restrict__ gamma,
                    const float* __restrict__ beta, unsigned short* __restrict__ tn) {
    __shared__ float s1[256];
    __shared__ float s2[256];
    int b   = blockIdx.x >> 6;
    int grp = blockIdx.x & 63;
    const float* tp = t + ((long)b * 256 + grp * 4) * 1024;
    float ls = 0.f, lq = 0.f;
    for (int i = threadIdx.x; i < 4096; i += 256) {
        float xv = tp[i]; ls += xv; lq += xv * xv;
    }
    s1[threadIdx.x] = ls; s2[threadIdx.x] = lq;
    __syncthreads();
    for (int off = 128; off > 0; off >>= 1) {
        if ((int)threadIdx.x < off) {
            s1[threadIdx.x] += s1[threadIdx.x + off];
            s2[threadIdx.x] += s2[threadIdx.x + off];
        }
        __syncthreads();
    }
    float mu  = s1[0] * (1.f / 4096.f);
    float var = s2[0] * (1.f / 4096.f) - mu * mu;
    float rs  = rsqrtf(var + 1e-5f);
    unsigned short* op = tn + ((long)b * 64 + grp) * 4096;
    for (int i = threadIdx.x; i < 4096; i += 256) {
        int r = i >> 10;
        float y = (tp[i] - mu) * rs * gamma[grp * 4 + r] + beta[grp * 4 + r];
        op[i] = f32_to_bf16(fmaxf(y, 0.f));
    }
}

// ------------------------------------------------- dynamic-conv aggregation
// out[b, g*64 + c*4 + r, h, w] = sum_{i,j} rot90^r(wmap)[b,g,(i,j),r,h,w] * v[b, g*64+c*4+r, h+i-3, w+j-3]
__global__ __launch_bounds__(256)
void dynconv_kernel(const float* __restrict__ wmap, const float* __restrict__ v,
                    float* __restrict__ out) {
    int tid = blockIdx.x * 256 + threadIdx.x;   // B*G*4*1024 = 262144
    int hw = tid & 1023;
    int r  = (tid >> 10) & 3;
    int g  = (tid >> 12) & 7;
    int b  = tid >> 15;
    int h = hw >> 5, w = hw & 31;

    const float* wm = wmap + (long)b * 392 * 4096 + (long)g * 49 * 4096 + r * 1024 + hw;
    const float* vb = v + ((long)b * 512 + g * 64 + r) * 1024;

    float acc[16];
#pragma unroll
    for (int c = 0; c < 16; ++c) acc[c] = 0.f;

    for (int i = 0; i < 7; ++i) {
        for (int j = 0; j < 7; ++j) {
            int y = h + i - 3, x = w + j - 3;
            if (y < 0 || y > 31 || x < 0 || x > 31) continue;
            int i2, j2;  // rot90^r source index (out[i,j] = in[i2,j2])
            if      (r == 0) { i2 = i;     j2 = j;     }
            else if (r == 1) { i2 = j;     j2 = 6 - i; }
            else if (r == 2) { i2 = 6 - i; j2 = 6 - j; }
            else             { i2 = 6 - j; j2 = i;     }
            float wv = wm[(i2 * 7 + j2) * 4096];
            const float* vp = vb + y * 32 + x;
#pragma unroll
            for (int c = 0; c < 16; ++c)
                acc[c] = fmaf(wv, vp[c * 4096], acc[c]);
        }
    }
    float* op = out + ((long)b * 512 + g * 64 + r) * 1024 + hw;
#pragma unroll
    for (int c = 0; c < 16; ++c) op[c * 4096] = acc[c];
}

// ---------------------------------------------------------------- launch
extern "C" void kernel_launch(void* const* d_in, const int* in_sizes, int n_in,
                              void* d_out, int out_size, void* d_ws, size_t ws_size,
                              hipStream_t stream) {
    const float* x    = (const float*)d_in[0];   // [8,512,32,32]
    const float* v_w  = (const float*)d_in[1];   // [128,128,4]
    const float* c1_w = (const float*)d_in[2];   // [64,128,4]
    const float* gn_g = (const float*)d_in[3];   // [256]
    const float* gn_b = (const float*)d_in[4];   // [256]
    const float* c2_w = (const float*)d_in[5];   // [392,64]
    const float* c2_b = (const float*)d_in[6];   // [392]
    float* out = (float*)d_out;                  // [8,512,32,32]

    char* ws = (char*)d_ws;
    unsigned short* xb   = (unsigned short*)ws;  ws += 8u  * 1024 * 1024;  // x bf16   [8,512,1024]
    unsigned short* w1l  = (unsigned short*)ws;  ws += 512 * 1024;         // lifted c1 [256,512]
    unsigned short* wvl  = (unsigned short*)ws;  ws += 1024 * 1024;        // lifted v_w [512,512]
    unsigned short* c2b  = (unsigned short*)ws;  ws += 64  * 1024;         // c2_w bf16 [392,64]
    float*          t    = (float*)ws;           ws += 8u  * 1024 * 1024;  // conv1 out [8,256,1024]
    unsigned short* tn   = (unsigned short*)ws;  ws += 4u  * 1024 * 1024;  // GN out bf16 [8,64,4096]
    float*          wmp  = (float*)ws;           ws += 52u * 1024 * 1024;  // wmap [8,392,4096]
    float*          vbuf = (float*)ws;           /* 16 MB */               // v [8,512,1024]

    // converts + weight lifting
    cvt_bf16_kernel<<<16384, 256, 0, stream>>>(x, xb, 4194304);
    cvt_bf16_kernel<<<98,    256, 0, stream>>>(c2_w, c2b, 25088);
    lift_kernel<<<512,  256, 0, stream>>>(c1_w, w1l, 131072);
    lift_kernel<<<1024, 256, 0, stream>>>(v_w,  wvl, 262144);

    // conv1: [256x512]x[512x1024] per batch -> t        (8*16*4 blocks)
    wmma_gemm_bf16<<<512, 256, 0, stream>>>(w1l, xb, t, nullptr,
        256, 16, 64, 512, 512, 512L * 1024, 256L * 1024);
    // GroupNorm + ReLU -> tn (bf16)
    gn_relu_kernel<<<512, 256, 0, stream>>>(t, gn_g, gn_b, tn);
    // conv2: [392x64]x[64x4096] per batch (+bias) -> wmap   (8*25*16 blocks)
    wmma_gemm_bf16<<<3200, 256, 0, stream>>>(c2b, tn, wmp, c2_b,
        392, 25, 256, 64, 64, 64L * 4096, 392L * 4096);
    // v branch: [512x512]x[512x1024] per batch -> vbuf      (8*32*4 blocks)
    wmma_gemm_bf16<<<1024, 256, 0, stream>>>(wvl, xb, vbuf, nullptr,
        512, 32, 64, 512, 512, 512L * 1024, 512L * 1024);
    // dynamic conv aggregation -> out
    dynconv_kernel<<<1024, 256, 0, stream>>>(wmp, vbuf, out);
}
